// MixtureOfExperts_38585986187450
// MI455X (gfx1250) — matrix operand order repaired and checked by reading, not compiled, not run on hardware
//
#include <hip/hip_runtime.h>
#include <hip/hip_bf16.h>
#include <math.h>

#define T_TOK 4096
#define DIMK  1024
#define NEXP  8
#define HID   2730
#define NCHUNK 22          // H padded to 22*128 = 2816
#define CHUNK 128
#define MT    16           // tokens per tile
#define KSTAGE 64          // K rows staged per phase-A iteration

typedef __attribute__((ext_vector_type(16))) __bf16         v16bf;
typedef __attribute__((ext_vector_type(16))) unsigned short v16u;
typedef __attribute__((ext_vector_type(8)))  unsigned short v8u;
typedef __attribute__((ext_vector_type(8)))  float          v8f;
typedef __attribute__((ext_vector_type(4)))  float          v4f;
typedef __attribute__((ext_vector_type(2)))  unsigned       v2ui;
typedef __attribute__((ext_vector_type(4)))  unsigned       v4ui;
// possibly-unaligned float4 (w_gate/w_up rows have stride 2730 floats)
typedef __attribute__((ext_vector_type(4), aligned(4))) float v4fu;

static __device__ __forceinline__ unsigned short f2bf(float f) {
  union { float f; unsigned u; } v; v.f = f;
  unsigned u = v.u;
  unsigned r = 0x7FFFu + ((u >> 16) & 1u);   // round-to-nearest-even
  return (unsigned short)((u + r) >> 16);
}

// pack two f32 -> two bf16 in one dword.
// Preferred: cvt_pk_bf16_f32 builtin; else one v_perm_b32 after +0x8000 round bias (fast-bf16 idiom).
static __device__ __forceinline__ unsigned pack2bf(float a, float b) {
#if __has_builtin(__builtin_amdgcn_cvt_pk_bf16_f32)
  auto p = __builtin_amdgcn_cvt_pk_bf16_f32(a, b);
  return __builtin_bit_cast(unsigned, p);
#elif __has_builtin(__builtin_amdgcn_perm)
  unsigned ua = __builtin_bit_cast(unsigned, a) + 0x8000u;
  unsigned ub = __builtin_bit_cast(unsigned, b) + 0x8000u;
  // bytes 0-1 from ua[31:16], bytes 2-3 from ub[31:16]
  return __builtin_amdgcn_perm(ub, ua, 0x07060302u);
#else
  return (unsigned)f2bf(a) | ((unsigned)f2bf(b) << 16);
#endif
}

// two contiguous 8-half LDS groups -> one v16bf WMMA operand (2x ds_load_b128)
static __device__ __forceinline__ v16bf frag_pair(const unsigned short* p0,
                                                  const unsigned short* p1) {
  v8u lo = *(const v8u*)p0;
  v8u hi = *(const v8u*)p1;
  v16u a = __builtin_shufflevector(lo, hi, 0, 1, 2, 3, 4, 5, 6, 7,
                                   8, 9, 10, 11, 12, 13, 14, 15);
  return __builtin_bit_cast(v16bf, a);
}

// ---------------------------------------------------------------------------
// 1) Router: logits, top-2 select + softmax weights, full probs, logsumexp^2
// ---------------------------------------------------------------------------
__global__ void router_kernel(const float* __restrict__ x,
                              const float* __restrict__ rw,
                              int* __restrict__ sel, float* __restrict__ selw,
                              float* __restrict__ probs, float* __restrict__ lse2)
{
  int t = blockIdx.x * blockDim.x + threadIdx.x;
  if (t >= T_TOK) return;
  float l[NEXP];
  #pragma unroll
  for (int e = 0; e < NEXP; ++e) l[e] = 0.0f;
  const float* xr = x + (size_t)t * DIMK;
  for (int d = 0; d < DIMK; ++d) {
    float xv = xr[d];
    const float* r = rw + (size_t)d * NEXP;
    #pragma unroll
    for (int e = 0; e < NEXP; ++e) l[e] += xv * r[e];
  }
  int i1 = 0;
  #pragma unroll
  for (int e = 1; e < NEXP; ++e) if (l[e] > l[i1]) i1 = e;
  int i2 = (i1 == 0) ? 1 : 0;
  #pragma unroll
  for (int e = 0; e < NEXP; ++e) if (e != i1 && l[e] > l[i2]) i2 = e;

  float a = l[i1], b = l[i2];
  float w2 = 1.0f / (1.0f + __expf(a - b));
  float w1 = 1.0f - w2;
  sel[t * 2 + 0] = i1; sel[t * 2 + 1] = i2;
  selw[t * 2 + 0] = w1; selw[t * 2 + 1] = w2;

  float m = a;
  float se = 0.0f;
  #pragma unroll
  for (int e = 0; e < NEXP; ++e) se += __expf(l[e] - m);
  float inv = 1.0f / se;
  #pragma unroll
  for (int e = 0; e < NEXP; ++e) probs[(size_t)t * NEXP + e] = __expf(l[e] - m) * inv;
  float lse = m + __logf(se);
  lse2[t] = lse * lse;
}

// ---------------------------------------------------------------------------
// 2) Deterministic per-expert token lists (one thread per expert, in order)
// ---------------------------------------------------------------------------
__global__ void build_lists_kernel(const int* __restrict__ sel,
                                   const float* __restrict__ selw,
                                   int* __restrict__ counts,
                                   int* __restrict__ tok_list,
                                   int* __restrict__ slot_list,
                                   float* __restrict__ wt_list)
{
  int e = threadIdx.x;
  if (e >= NEXP) return;
  int c = 0;
  for (int t = 0; t < T_TOK; ++t) {
    #pragma unroll
    for (int k = 0; k < 2; ++k) {
      if (sel[t * 2 + k] == e) {
        tok_list[e * T_TOK + c]  = t;
        slot_list[e * T_TOK + c] = k;
        wt_list[e * T_TOK + c]   = selw[t * 2 + k];
        ++c;
      }
    }
  }
  counts[e] = c;
}

// ---------------------------------------------------------------------------
// 3) Fused SwiGLU expert, bf16 WMMA, transposed-LDS B staging,
//    software-pipelined phase-A weight staging.
//    Dynamic LDS layout (halfwords):
//      Xt  [MT*DIMK]        32 KB  bf16 X tile, row-major [m][k]
//      Hc  [MT*CHUNK]        4 KB  bf16 h chunk, row-major [m][k]
//      BgT [CHUNK*KSTAGE]   16 KB  gate B, transposed [n][k]
//      BuT [CHUNK*KSTAGE]   16 KB  up   B, transposed [n][k]
//      BdT [DIMK*32]        64 KB  down B, transposed [n][k]
// ---------------------------------------------------------------------------
#define SMEM_HALFWORDS (MT*DIMK + MT*CHUNK + 2*CHUNK*KSTAGE + DIMK*32)
#define SMEM_BYTES (SMEM_HALFWORDS * 2)

__global__ __launch_bounds__(256) void moe_expert_kernel(
    const float* __restrict__ x,  const float* __restrict__ wg,
    const float* __restrict__ wu, const float* __restrict__ wd,
    const int* __restrict__ counts, const int* __restrict__ tok_list,
    const int* __restrict__ slot_list, const float* __restrict__ wt_list,
    float* __restrict__ contrib)
{
  extern __shared__ unsigned short dynsmem[];
  unsigned short* Xt  = dynsmem;
  unsigned short* Hc  = Xt  + MT * DIMK;
  unsigned short* BgT = Hc  + MT * CHUNK;
  unsigned short* BuT = BgT + CHUNK * KSTAGE;
  unsigned short* BdT = BuT + CHUNK * KSTAGE;
  __shared__ int   tokS[MT];
  __shared__ int   slotS[MT];
  __shared__ float wtS[MT];

  const int e    = blockIdx.y;
  const int tile = blockIdx.x;
  const int cnt  = counts[e];
  const int base = tile * MT;
  if (base >= cnt) return;                     // uniform across block
  int mcnt = cnt - base; if (mcnt > MT) mcnt = MT;

  const int tid  = threadIdx.x;
  const int wave = tid >> 5;
  const int lane = tid & 31;
  const int lhi  = (lane >= 16) ? 1 : 0;
  const int l16  = lane & 15;

  if (tid < MT) {
    int idx = e * T_TOK + base + tid;
    int ok  = (tid < mcnt);
    tokS[tid]  = ok ? tok_list[idx]  : 0;
    slotS[tid] = ok ? slot_list[idx] : 0;
    wtS[tid]   = ok ? wt_list[idx]   : 0.0f;
  }
  __syncthreads();

  // stage gathered X tile as bf16 (2x b128 load, packed cvt, 1x b128 store)
  for (int i8 = tid; i8 < MT * DIMK / 8; i8 += 256) {
    int i = i8 * 8;
    int m = i >> 10;
    v4ui o;
    if (m < mcnt) {
      const v4f* src = (const v4f*)(x + (size_t)tokS[m] * DIMK + (i & (DIMK - 1)));
      v4f a = src[0], b = src[1];
      o[0] = pack2bf(a[0], a[1]); o[1] = pack2bf(a[2], a[3]);
      o[2] = pack2bf(b[0], b[1]); o[3] = pack2bf(b[2], b[3]);
    } else {
      #pragma unroll
      for (int j = 0; j < 4; ++j) o[j] = 0u;
    }
    *(v4ui*)&Xt[i] = o;
  }
  __syncthreads();

  // down-GEMM accumulators: wave owns output cols [wave*128, wave*128+128)
  v8f accD[8];
  #pragma unroll
  for (int i = 0; i < 8; ++i)
    #pragma unroll
    for (int j = 0; j < 8; ++j) accD[i][j] = 0.0f;

  for (int hc = 0; hc < NCHUNK; ++hc) {
    const bool full = (hc + 1) * CHUNK <= HID;   // only hc==21 is partial
    const int  nbase = hc * CHUNK;

    // ---- phase A: gate/up GEMM for this 128-col chunk (pipelined) ----
    v8f accG, accU;
    #pragma unroll
    for (int j = 0; j < 8; ++j) { accG[j] = 0.0f; accU[j] = 0.0f; }

    // in-flight slab registers: 2 micro-tiles (4k x 4n) per matrix per thread
    v4fu rg[2][4], ru[2][4];

    auto loadA = [&](int ko) {
      #pragma unroll
      for (int hlf = 0; hlf < 2; ++hlf) {
        int mt = tid + hlf * 256;
        int k0 = (mt >> 5) * 4;
        int n0 = (mt & 31) * 4;
        int gk = ko * KSTAGE + k0;
        int gn = nbase + n0;
        if (full) {
          #pragma unroll
          for (int r = 0; r < 4; ++r) {
            size_t off = ((size_t)e * DIMK + gk + r) * HID + gn;
            rg[hlf][r] = *(const v4fu*)(wg + off);
            ru[hlf][r] = *(const v4fu*)(wu + off);
          }
        } else {
          #pragma unroll
          for (int r = 0; r < 4; ++r) {
            size_t off = ((size_t)e * DIMK + gk + r) * HID + gn;
            #pragma unroll
            for (int c = 0; c < 4; ++c) {
              int ok = (gn + c) < HID;
              rg[hlf][r][c] = ok ? wg[off + c] : 0.0f;
              ru[hlf][r][c] = ok ? wu[off + c] : 0.0f;
            }
          }
        }
      }
    };
    auto storeA = [&]() {
      #pragma unroll
      for (int hlf = 0; hlf < 2; ++hlf) {
        int mt = tid + hlf * 256;
        int k0 = (mt >> 5) * 4;
        int n0 = (mt & 31) * 4;
        #pragma unroll
        for (int c = 0; c < 4; ++c) {
          v2ui cg, cu;
          cg[0] = pack2bf(rg[hlf][0][c], rg[hlf][1][c]);
          cg[1] = pack2bf(rg[hlf][2][c], rg[hlf][3][c]);
          cu[0] = pack2bf(ru[hlf][0][c], ru[hlf][1][c]);
          cu[1] = pack2bf(ru[hlf][2][c], ru[hlf][3][c]);
          *(v2ui*)&BgT[(n0 + c) * KSTAGE + k0] = cg;
          *(v2ui*)&BuT[(n0 + c) * KSTAGE + k0] = cu;
        }
      }
    };

    loadA(0);
    storeA();
    __syncthreads();

    for (int ko = 0; ko < DIMK / KSTAGE; ++ko) {
      // issue next slab's global loads before computing this slab
      if (ko + 1 < DIMK / KSTAGE) loadA(ko + 1);

      #pragma unroll
      for (int kk = 0; kk < KSTAGE / 32; ++kk) {
        const unsigned short* ab = &Xt[l16 * DIMK + ko * KSTAGE + kk * 32];
        v16bf af = frag_pair(ab + lhi * 16, ab + lhi * 16 + 8);
        const unsigned short* gb = &BgT[(wave * 16 + l16) * KSTAGE + kk * 32 + lhi * 16];
        v16bf bg = frag_pair(gb, gb + 8);
        const unsigned short* ub = &BuT[(wave * 16 + l16) * KSTAGE + kk * 32 + lhi * 16];
        v16bf bu = frag_pair(ub, ub + 8);
        accG = __builtin_amdgcn_wmma_f32_16x16x32_bf16(false, af, false, bg,
                                                       (short)0, accG, false, false);
        accU = __builtin_amdgcn_wmma_f32_16x16x32_bf16(false, af, false, bu,
                                                       (short)0, accU, false, false);
      }

      if (ko + 1 < DIMK / KSTAGE) {
        __syncthreads();   // slab ko fully consumed
        storeA();          // commit slab ko+1 to LDS
        __syncthreads();   // slab ko+1 visible
      }
    }

    // h = silu(gate) * up -> bf16 chunk in LDS (C/D layout -> rows/cols)
    #pragma unroll
    for (int r = 0; r < 8; ++r) {
      float g = accG[r], u = accU[r];
      float h = (g / (1.0f + __expf(-g))) * u;
      int m = r + (lhi ? 8 : 0);
      Hc[m * CHUNK + wave * 16 + l16] = f2bf(h);
    }
    __syncthreads();

    // ---- phase B: down GEMM over this chunk's 128 K rows ----
    for (int ks = 0; ks < CHUNK / 32; ++ks) {
      // stage W_down slab [32 x DIMK] transposed, 4x4 micro-tiles
      for (int mt = tid; mt < (32 / 4) * (DIMK / 4); mt += 256) {
        int k0 = (mt >> 8) * 4;
        int n0 = (mt & 255) * 4;
        int ghr = nbase + ks * 32 + k0;
        float td[4][4];
        #pragma unroll
        for (int r = 0; r < 4; ++r) {
          if (ghr + r < HID) {
            v4f v = *(const v4f*)(wd + ((size_t)e * HID + ghr + r) * DIMK + n0);
            #pragma unroll
            for (int c = 0; c < 4; ++c) td[r][c] = v[c];
          } else {
            #pragma unroll
            for (int c = 0; c < 4; ++c) td[r][c] = 0.0f;
          }
        }
        #pragma unroll
        for (int c = 0; c < 4; ++c) {
          v2ui cd;
          cd[0] = pack2bf(td[0][c], td[1][c]);
          cd[1] = pack2bf(td[2][c], td[3][c]);
          *(v2ui*)&BdT[(n0 + c) * 32 + k0] = cd;
        }
      }
      __syncthreads();

      // prefetch next slab of W_down while this one is being consumed
      if (ks + 1 < CHUNK / 32) {
        int ghr2 = nbase + (ks + 1) * 32 + (tid >> 3);
        if (ghr2 < HID)
          __builtin_prefetch(wd + ((size_t)e * HID + ghr2) * DIMK + (tid & 7) * 128, 0, 0);
      }

      const unsigned short* ab = &Hc[l16 * CHUNK + ks * 32];
      v16bf af = frag_pair(ab + lhi * 16, ab + lhi * 16 + 8);
      #pragma unroll
      for (int t = 0; t < 8; ++t) {
        const unsigned short* bb = &BdT[(wave * 128 + t * 16 + l16) * 32 + lhi * 16];
        v16bf bf = frag_pair(bb, bb + 8);
        accD[t] = __builtin_amdgcn_wmma_f32_16x16x32_bf16(false, af, false, bf,
                                                          (short)0, accD[t], false, false);
      }
      __syncthreads();   // slab consumed before restage / next chunk
    }
  }

  // epilogue: weighted scatter, one writer per (token, slot) row
  #pragma unroll
  for (int t = 0; t < 8; ++t) {
    #pragma unroll
    for (int r = 0; r < 8; ++r) {
      int m = r + (lhi ? 8 : 0);
      if (m < mcnt) {
        int tok = tokS[m], slot = slotS[m];
        float w = wtS[m];
        int col = wave * 128 + t * 16 + l16;
        contrib[((size_t)tok * 2 + slot) * DIMK + col] = w * accD[t][r];
      }
    }
  }
}

// ---------------------------------------------------------------------------
// 4) Combine the two expert contributions per token (fixed order)
// ---------------------------------------------------------------------------
__global__ void combine_kernel(const float* __restrict__ contrib,
                               float* __restrict__ out)
{
  int i = blockIdx.x * blockDim.x + threadIdx.x;
  if (i >= T_TOK * DIMK) return;
  int t = i >> 10; int d = i & (DIMK - 1);
  out[i] = contrib[((size_t)t * 2 + 0) * DIMK + d]
         + contrib[((size_t)t * 2 + 1) * DIMK + d];
}

// ---------------------------------------------------------------------------
// 5) Aux losses: deterministic tree reductions
// ---------------------------------------------------------------------------
__global__ void loss_kernel(const float* __restrict__ probs,
                            const float* __restrict__ lse2,
                            float* __restrict__ outS)
{
  __shared__ float red[256];
  __shared__ float usum[NEXP];
  int tid = threadIdx.x;
  float usage[NEXP];
  #pragma unroll
  for (int e = 0; e < NEXP; ++e) usage[e] = 0.0f;
  float ls = 0.0f;
  for (int t = tid; t < T_TOK; t += 256) {
    #pragma unroll
    for (int e = 0; e < NEXP; ++e) usage[e] += probs[(size_t)t * NEXP + e];
    ls += lse2[t];
  }
  for (int e = 0; e < NEXP; ++e) {
    red[tid] = usage[e]; __syncthreads();
    for (int s = 128; s > 0; s >>= 1) {
      if (tid < s) red[tid] += red[tid + s];
      __syncthreads();
    }
    if (tid == 0) usum[e] = red[0];
    __syncthreads();
  }
  red[tid] = ls; __syncthreads();
  for (int s = 128; s > 0; s >>= 1) {
    if (tid < s) red[tid] += red[tid + s];
    __syncthreads();
  }
  if (tid == 0) {
    float lb = 0.0f;
    #pragma unroll
    for (int e = 0; e < NEXP; ++e) {
      float u = usum[e] / (float)T_TOK;
      lb += u * u;
    }
    outS[0] = (float)NEXP * lb;
    outS[1] = red[0] / (float)T_TOK;
  }
}

// ---------------------------------------------------------------------------
extern "C" void kernel_launch(void* const* d_in, const int* in_sizes, int n_in,
                              void* d_out, int out_size, void* d_ws, size_t ws_size,
                              hipStream_t stream)
{
  (void)in_sizes; (void)n_in; (void)out_size; (void)ws_size;
  const float* x  = (const float*)d_in[0];
  const float* rw = (const float*)d_in[1];
  const float* wg = (const float*)d_in[2];
  const float* wu = (const float*)d_in[3];
  const float* wd = (const float*)d_in[4];
  float* out = (float*)d_out;

  char* p = (char*)d_ws;
  float* contrib = (float*)p; p += (size_t)T_TOK * 2 * DIMK * sizeof(float); // 32 MB
  int*   sel     = (int*)p;   p += (size_t)T_TOK * 2 * sizeof(int);
  float* selw    = (float*)p; p += (size_t)T_TOK * 2 * sizeof(float);
  float* probs   = (float*)p; p += (size_t)T_TOK * NEXP * sizeof(float);
  float* lse2    = (float*)p; p += (size_t)T_TOK * sizeof(float);
  int*   counts  = (int*)p;   p += 64;
  int*   tokl    = (int*)p;   p += (size_t)NEXP * T_TOK * sizeof(int);
  int*   slotl   = (int*)p;   p += (size_t)NEXP * T_TOK * sizeof(int);
  float* wtl     = (float*)p; p += (size_t)NEXP * T_TOK * sizeof(float);

  router_kernel<<<T_TOK / 256, 256, 0, stream>>>(x, rw, sel, selw, probs, lse2);
  build_lists_kernel<<<1, 64, 0, stream>>>(sel, selw, counts, tokl, slotl, wtl);

  hipFuncSetAttribute((const void*)moe_expert_kernel,
                      hipFuncAttributeMaxDynamicSharedMemorySize, SMEM_BYTES);
  dim3 grid(T_TOK / MT, NEXP);
  moe_expert_kernel<<<grid, 256, SMEM_BYTES, stream>>>(x, wg, wu, wd, counts,
                                                       tokl, slotl, wtl, contrib);

  combine_kernel<<<(T_TOK * DIMK) / 256, 256, 0, stream>>>(contrib, out);
  loss_kernel<<<1, 256, 0, stream>>>(probs, lse2, out + (size_t)T_TOK * DIMK);
}
